// ETEButterflyLayer_16192026705974
// MI455X (gfx1250) — matrix operand order repaired and checked by reading, not compiled, not run on hardware
//
#include <hip/hip_runtime.h>

typedef __bf16 bf16;
typedef __attribute__((ext_vector_type(16))) __bf16 v16bf;
typedef __attribute__((ext_vector_type(8)))  float  v8f;

union AFrag { v16bf v; uint4 u[2]; };

#define WMMA_BF16(A, B, C) \
  __builtin_amdgcn_wmma_f32_16x16x32_bf16(false, (A), false, (B), (short)0, (C), false, false)

// ---------------------------------------------------------------------------
// Encoder input conv: rows = B*512 = 131072; out[r,c] = relu(x[r,:16] . F[:,c] + b)
// ---------------------------------------------------------------------------
__global__ __launch_bounds__(256) void in_conv_enc(
    const float* __restrict__ x, const float* __restrict__ F,
    const float* __restrict__ bias, bf16* __restrict__ out, int nrows) {
  __shared__ float Fs[16 * 32];
  __shared__ float bs[32];
  int tid = threadIdx.x;
  for (int i = tid; i < 512; i += 256) Fs[i] = F[i];
  if (tid < 32) bs[tid] = bias[tid];
  __syncthreads();
  int r = blockIdx.x * 256 + tid;
  if (r >= nrows) return;
  const float* xp = x + (size_t)r * 16;
  float v[16];
#pragma unroll
  for (int f = 0; f < 16; ++f) v[f] = xp[f];
  bf16* op = out + (size_t)r * 32;
#pragma unroll
  for (int c = 0; c < 32; ++c) {
    float acc = bs[c];
#pragma unroll
    for (int f = 0; f < 16; ++f) acc += v[f] * Fs[f * 32 + c];
    op[c] = (bf16)(acc > 0.f ? acc : 0.f);
  }
}

// ---------------------------------------------------------------------------
// Decoder input conv: rows = 2B*512 = 262144.
// dx[r,f] = mid[b*16384 + l*32 + 2f + ri], r = (ri*256+b)*512 + l   (mult pre-applied)
// ---------------------------------------------------------------------------
__global__ __launch_bounds__(256) void in_conv_dec(
    const float* __restrict__ mid, const float* __restrict__ F,
    const float* __restrict__ bias, bf16* __restrict__ out, int nrows) {
  __shared__ float Fs[16 * 32];
  __shared__ float bs[32];
  int tid = threadIdx.x;
  for (int i = tid; i < 512; i += 256) Fs[i] = F[i];
  if (tid < 32) bs[tid] = bias[tid];
  __syncthreads();
  int r = blockIdx.x * 256 + tid;
  if (r >= nrows) return;
  int i = r >> 9, l = r & 511;
  int ri = i >> 8, b = i & 255;
  const float* mp = mid + (size_t)b * 16384 + (size_t)l * 32 + ri;
  float v[16];
#pragma unroll
  for (int f = 0; f < 16; ++f) v[f] = mp[2 * f];
  bf16* op = out + (size_t)r * 32;
#pragma unroll
  for (int c = 0; c < 32; ++c) {
    float acc = bs[c];
#pragma unroll
    for (int f = 0; f < 16; ++f) acc += v[f] * Fs[f * 32 + c];
    op[c] = (bf16)(acc > 0.f ? acc : 0.f);
  }
}

// ---------------------------------------------------------------------------
// Tree level: per branch k_out, GEMM A[Mk,64](bf16, contiguous) x W2[64,32] + bias, relu.
//   W flat (Kout, 2, 32, 32) == per-branch row-major (64, 32).
//   Input branch base = (k>>1)*Mk*64 elements; row m reads 64 contiguous bf16 at m*64.
// Each wave: TWO 16-row tiles per iteration -> 8 WMMAs across 4 independent
// accumulator chains (hides the bf16 WMMA->WMMA RAW hazard), 2KB coalesced store.
// ---------------------------------------------------------------------------
__global__ __launch_bounds__(256) void tree_level_kernel(
    const bf16* __restrict__ sin, bf16* __restrict__ sout,
    const float* __restrict__ W, const float* __restrict__ bias,
    int Brows, int Lout) {
  __shared__ alignas(32) bf16 Wt[64 * 32];       // transposed: Wt[d*64 + j] = W2[j, d]
  __shared__ alignas(16) bf16 OutT[8 * 32 * 32]; // 2 tiles per wave
  const int k = blockIdx.y;
  const int tid = threadIdx.x;
  const float* Wk = W + (size_t)k * 2048;
  for (int idx = tid; idx < 2048; idx += 256) {
    int d = idx >> 6, j = idx & 63;
    Wt[idx] = (bf16)Wk[j * 32 + d];
  }
  __syncthreads();

  const int lane = tid & 31, wave = tid >> 5;
  const int hi = lane >> 4, n = lane & 15;

  // B fragments: lane = column (ct*16+n); lanes 0-15 hold K s*32+0..15, lanes 16-31 K s*32+16..31
  v16bf bfr[2][2];
#pragma unroll
  for (int ct = 0; ct < 2; ++ct)
#pragma unroll
    for (int s = 0; s < 2; ++s)
      bfr[ct][s] = *(const v16bf*)&Wt[(ct * 16 + n) * 64 + s * 32 + hi * 16];

  const float b0 = bias[k * 32 + n];
  const float b1 = bias[k * 32 + 16 + n];
  const int Mk = Brows * Lout;
  const int tilePairs = Mk >> 5;                 // 32 rows per iteration
  const bf16* Ain = sin + (size_t)(k >> 1) * Mk * 64;
  bf16* Obr = sout + (size_t)k * Mk * 32;
  bf16* myOut = &OutT[wave * 1024];
  const int stride = gridDim.x * 8;
  const size_t strideElems = (size_t)stride * 32 * 64;

  for (int tp = blockIdx.x * 8 + wave; tp < tilePairs; tp += stride) {
    const bf16* rowp0 = Ain + (size_t)(tp * 32 + n) * 64;       // tile 0
    const bf16* rowp1 = rowp0 + 16 * 64;                        // tile 1
    // prefetch next iteration's A stream (speculative; OOB is dropped)
    __builtin_prefetch(rowp0 + strideElems, 0, 1);
    __builtin_prefetch(rowp1 + strideElems, 0, 1);

    AFrag a0, a1, a2, a3;  // K-steps 0/1 for tiles 0/1, contiguous k = 16*hi + s, same as B
    a0.u[0] = *(const uint4*)(rowp0 + hi * 16);
    a0.u[1] = *(const uint4*)(rowp0 + 8 + hi * 16);
    a1.u[0] = *(const uint4*)(rowp0 + 32 + hi * 16);
    a1.u[1] = *(const uint4*)(rowp0 + 40 + hi * 16);
    a2.u[0] = *(const uint4*)(rowp1 + hi * 16);
    a2.u[1] = *(const uint4*)(rowp1 + 8 + hi * 16);
    a3.u[0] = *(const uint4*)(rowp1 + 32 + hi * 16);
    a3.u[1] = *(const uint4*)(rowp1 + 40 + hi * 16);

    v8f p0 = {b0, b0, b0, b0, b0, b0, b0, b0};
    v8f p1 = {b1, b1, b1, b1, b1, b1, b1, b1};
    v8f q0 = p0, q1 = p1;
    // 4 independent accumulator chains -> no dependent back-to-back WMMAs
    p0 = WMMA_BF16(a0.v, bfr[0][0], p0);
    p1 = WMMA_BF16(a0.v, bfr[1][0], p1);
    q0 = WMMA_BF16(a2.v, bfr[0][0], q0);
    q1 = WMMA_BF16(a2.v, bfr[1][0], q1);
    p0 = WMMA_BF16(a1.v, bfr[0][1], p0);
    p1 = WMMA_BF16(a1.v, bfr[1][1], p1);
    q0 = WMMA_BF16(a3.v, bfr[0][1], q0);
    q1 = WMMA_BF16(a3.v, bfr[1][1], q1);

#pragma unroll
    for (int j = 0; j < 8; ++j) {
      int row = hi * 8 + j;
      float r0 = p0[j] > 0.f ? p0[j] : 0.f;
      float r1 = p1[j] > 0.f ? p1[j] : 0.f;
      float r2 = q0[j] > 0.f ? q0[j] : 0.f;
      float r3 = q1[j] > 0.f ? q1[j] : 0.f;
      myOut[row * 32 + n]              = (bf16)r0;
      myOut[row * 32 + 16 + n]         = (bf16)r1;
      myOut[512 + row * 32 + n]        = (bf16)r2;
      myOut[512 + row * 32 + 16 + n]   = (bf16)r3;
    }
    asm volatile("s_wait_dscnt 0" ::: "memory");
    // 32x32 bf16 tile-pair == 2048 contiguous bytes at Obr + tp*1024 elements
    const uint4* src = (const uint4*)myOut;
    uint4* dst = (uint4*)(Obr + (size_t)tp * 1024);
#pragma unroll
    for (int c = 0; c < 4; ++c) dst[lane * 4 + c] = src[lane * 4 + c];
  }
}

// ---------------------------------------------------------------------------
// fea_dense + mid_dense: per branch k: leaves[k*256 + b, 0..31] x Wfea[k](32x32),
// scale by mult[k*32+c], store fp32 to mid[b*16384 + k*32 + c].
// ---------------------------------------------------------------------------
__global__ __launch_bounds__(256) void fea_mid_kernel(
    const bf16* __restrict__ leaves, const float* __restrict__ Wfea,
    const float* __restrict__ mult, float* __restrict__ mid) {
  __shared__ alignas(32) bf16 Wt[32 * 32];      // Wt[d*32 + c] = W[c, d]
  __shared__ alignas(16) float OutT[8 * 16 * 32];
  const int k = blockIdx.y;
  const int tid = threadIdx.x;
  const float* Wk = Wfea + (size_t)k * 1024;
  for (int idx = tid; idx < 1024; idx += 256) {
    int d = idx >> 5, c = idx & 31;
    Wt[idx] = (bf16)Wk[c * 32 + d];
  }
  __syncthreads();
  const int lane = tid & 31, wave = tid >> 5;
  const int hi = lane >> 4, n = lane & 15;
  v16bf bf0 = *(const v16bf*)&Wt[n * 32 + hi * 16];
  v16bf bf1 = *(const v16bf*)&Wt[(16 + n) * 32 + hi * 16];
  const float m0 = mult[k * 32 + n];
  const float m1 = mult[k * 32 + 16 + n];
  const bf16* Ak = leaves + (size_t)k * 256 * 32;
  float* myOut = &OutT[wave * 512];
  const int tiles = 256 >> 4;
  for (int t = blockIdx.x * 8 + wave; t < tiles; t += gridDim.x * 8) {
    const bf16* rowp = Ak + (size_t)(t * 16 + n) * 32;
    AFrag a;
    a.u[0] = *(const uint4*)(rowp + hi * 16);
    a.u[1] = *(const uint4*)(rowp + 8 + hi * 16);
    v8f acc0 = {0, 0, 0, 0, 0, 0, 0, 0};
    v8f acc1 = {0, 0, 0, 0, 0, 0, 0, 0};
    acc0 = WMMA_BF16(a.v, bf0, acc0);
    acc1 = WMMA_BF16(a.v, bf1, acc1);
#pragma unroll
    for (int j = 0; j < 8; ++j) {
      int row = hi * 8 + j;
      myOut[row * 32 + n]      = acc0[j] * m0;
      myOut[row * 32 + 16 + n] = acc1[j] * m1;
    }
    asm volatile("s_wait_dscnt 0" ::: "memory");
    int r = lane >> 1, h = lane & 1;   // 64B per lane, contiguous per output row
    const float* src = &myOut[r * 32 + h * 16];
    float* dst = &mid[(size_t)(t * 16 + r) * 16384 + k * 32 + h * 16];
    *(uint4*)(dst + 0)  = *(const uint4*)(src + 0);
    *(uint4*)(dst + 4)  = *(const uint4*)(src + 4);
    *(uint4*)(dst + 8)  = *(const uint4*)(src + 8);
    *(uint4*)(dst + 12) = *(const uint4*)(src + 12);
  }
}

// ---------------------------------------------------------------------------
// Final: per branch k compute O[k, i, :] = leaves[k*512 + i] x Wfea[k] for
// i = b and i = 256+b, then out[b, k*16+u] = O[k,b,2u] - O[k,256+b,2u+1].
// ---------------------------------------------------------------------------
__global__ __launch_bounds__(256) void final_kernel(
    const bf16* __restrict__ leaves, const float* __restrict__ Wfea,
    float* __restrict__ out) {
  __shared__ alignas(32) bf16 Wt[32 * 32];
  __shared__ alignas(16) float T[8 * 2 * 512];
  const int k = blockIdx.y;
  const int tid = threadIdx.x;
  const float* Wk = Wfea + (size_t)k * 1024;
  for (int idx = tid; idx < 1024; idx += 256) {
    int d = idx >> 5, c = idx & 31;
    Wt[idx] = (bf16)Wk[c * 32 + d];
  }
  __syncthreads();
  const int lane = tid & 31, wave = tid >> 5;
  const int hi = lane >> 4, n = lane & 15;
  v16bf bf0 = *(const v16bf*)&Wt[n * 32 + hi * 16];
  v16bf bf1 = *(const v16bf*)&Wt[(16 + n) * 32 + hi * 16];
  const bf16* Ak = leaves + (size_t)k * 512 * 32;
  float* T0 = &T[wave * 1024];
  float* T1 = T0 + 512;
  const int tiles = 256 >> 4;
  for (int t = blockIdx.x * 8 + wave; t < tiles; t += gridDim.x * 8) {
    const bf16* r0p = Ak + (size_t)(t * 16 + n) * 32;        // rows b
    const bf16* r1p = Ak + (size_t)(256 + t * 16 + n) * 32;  // rows 256+b
    AFrag a0, a1;
    a0.u[0] = *(const uint4*)(r0p + hi * 16);
    a0.u[1] = *(const uint4*)(r0p + 8 + hi * 16);
    a1.u[0] = *(const uint4*)(r1p + hi * 16);
    a1.u[1] = *(const uint4*)(r1p + 8 + hi * 16);
    v8f c00 = {0, 0, 0, 0, 0, 0, 0, 0}, c01 = c00, c10 = c00, c11 = c00;
    c00 = WMMA_BF16(a0.v, bf0, c00);
    c01 = WMMA_BF16(a0.v, bf1, c01);
    c10 = WMMA_BF16(a1.v, bf0, c10);
    c11 = WMMA_BF16(a1.v, bf1, c11);
#pragma unroll
    for (int j = 0; j < 8; ++j) {
      int row = hi * 8 + j;
      T0[row * 32 + n]      = c00[j];
      T0[row * 32 + 16 + n] = c01[j];
      T1[row * 32 + n]      = c10[j];
      T1[row * 32 + 16 + n] = c11[j];
    }
    asm volatile("s_wait_dscnt 0" ::: "memory");
#pragma unroll
    for (int j = 0; j < 8; ++j) {
      int i = hi * 8 + j;
      float r = T0[i * 32 + 2 * n] - T1[i * 32 + 2 * n + 1];
      out[(size_t)(t * 16 + i) * 8192 + k * 16 + n] = r;
    }
    asm volatile("s_wait_dscnt 0" ::: "memory");
  }
}

// ---------------------------------------------------------------------------
extern "C" void kernel_launch(void* const* d_in, const int* in_sizes, int n_in,
                              void* d_out, int out_size, void* d_ws, size_t ws_size,
                              hipStream_t stream) {
  (void)in_sizes; (void)n_in; (void)out_size; (void)ws_size;
  const float* in_data      = (const float*)d_in[0];
  const float* en_in_filter = (const float*)d_in[1];
  const float* en_in_bias   = (const float*)d_in[2];
  const float* en_filters[9];
  const float* en_biases[9];
  for (int l = 0; l < 9; ++l) { en_filters[l] = (const float*)d_in[3 + l];
                                en_biases[l]  = (const float*)d_in[12 + l]; }
  const float* en_fea       = (const float*)d_in[21];
  const float* mid_dense    = (const float*)d_in[22];
  const float* de_in_filter = (const float*)d_in[23];
  const float* de_in_bias   = (const float*)d_in[24];
  const float* de_filters[9];
  const float* de_biases[9];
  for (int l = 0; l < 9; ++l) { de_filters[l] = (const float*)d_in[25 + l];
                                de_biases[l]  = (const float*)d_in[34 + l]; }
  const float* de_fea = (const float*)d_in[43];
  float* out = (float*)d_out;

  char* ws = (char*)d_ws;
  bf16*  sA  = (bf16*)ws;                           // 16 MiB bf16 state
  bf16*  sB  = (bf16*)(ws + (size_t)(16u << 20));   // 16 MiB bf16 state
  float* mid = (float*)(ws + (size_t)(32u << 20));  // 16 MiB fp32 mid

  // Encoder input conv: (B,512,16) x (16,32) -> bf16 state [1,256,512,32]
  in_conv_enc<<<512, 256, 0, stream>>>(in_data, en_in_filter, en_in_bias, sA, 131072);

  bf16 *cur = sA, *nxt = sB;
  for (int l = 0; l < 9; ++l) {
    int Kout = 1 << (l + 1);
    int Lout = 512 >> (l + 1);
    int tilePairs = (256 * Lout) >> 5;
    int gx = tilePairs >> 3; if (gx < 1) gx = 1; if (gx > 512) gx = 512;
    tree_level_kernel<<<dim3(gx, Kout), 256, 0, stream>>>(
        cur, nxt, en_filters[l], en_biases[l], 256, Lout);
    bf16* tmp = cur; cur = nxt; nxt = tmp;
  }
  // cur == leaves (512, 256, 32)

  fea_mid_kernel<<<dim3(2, 512), 256, 0, stream>>>(cur, en_fea, mid_dense, mid);

  // Decoder input conv -> bf16 state [1,512,512,32] (reuse sA; leaves already consumed)
  in_conv_dec<<<1024, 256, 0, stream>>>(mid, de_in_filter, de_in_bias, sA, 262144);

  cur = sA; nxt = sB;
  for (int l = 0; l < 9; ++l) {
    int Kout = 1 << (l + 1);
    int Lout = 512 >> (l + 1);
    int tilePairs = (512 * Lout) >> 5;
    int gx = tilePairs >> 3; if (gx < 1) gx = 1; if (gx > 512) gx = 512;
    tree_level_kernel<<<dim3(gx, Kout), 256, 0, stream>>>(
        cur, nxt, de_filters[l], de_biases[l], 512, Lout);
    bf16* tmp = cur; cur = nxt; nxt = tmp;
  }
  // cur == decoder leaves (512, 512, 32)

  final_kernel<<<dim3(2, 512), 256, 0, stream>>>(cur, de_fea, out);
}
